// AttentionHead_14688788152482
// MI455X (gfx1250) — matrix operand order repaired and checked by reading, not compiled, not run on hardware
//
#include <hip/hip_runtime.h>

// ---------------------------------------------------------------------------
// CDNA5 (gfx1250) bf16 WMMA attention:
//   Phase 1: Q/K/V = x @ W{q,k,v}  (f32 -> bf16 via v_cvt_pk_bf16_f32,
//            v_wmma_f32_16x16x32_bf16, double-buffered LDS + reg prefetch)
//   Phase 2: causal flash attention, online softmax, bf16 WMMA for S and PV,
//            double-buffered K/V tiles; K tile copied with
//            GLOBAL_LOAD_ASYNC_TO_LDS_B128 (ASYNCcnt path).
// ---------------------------------------------------------------------------

typedef __attribute__((ext_vector_type(16))) __bf16 bf16x16;
typedef __attribute__((ext_vector_type(2)))  __bf16 bf16x2;
typedef __attribute__((ext_vector_type(8)))  float  f32x8;
typedef int v4i __attribute__((vector_size(16)));   // matches async builtin param

union ABFrag { unsigned u[8]; bf16x16 v; };
union CFrag  { float    f[8]; f32x8   v; };

// --- f32 -> bf16 conversion: single-instruction pack when available --------
#if defined(__has_builtin)
#  if __has_builtin(__builtin_amdgcn_cvt_pk_bf16_f32)
#    define USE_CVT_PK_BF16 1
#  endif
#endif
#ifndef USE_CVT_PK_BF16
#  define USE_CVT_PK_BF16 0
#endif

__device__ __forceinline__ unsigned bf16_rne_bits(float x) {
  union { float f; unsigned u; } c; c.f = x;
  return (c.u + 0x7FFFu + ((c.u >> 16) & 1u)) >> 16;
}
__device__ __forceinline__ unsigned pack_bf16x2(float lo, float hi) {
#if USE_CVT_PK_BF16
  union { bf16x2 v; unsigned u; } c;
  c.v = __builtin_amdgcn_cvt_pk_bf16_f32(lo, hi);
  return c.u;
#else
  return bf16_rne_bits(lo) | (bf16_rne_bits(hi) << 16);
#endif
}
__device__ __forceinline__ unsigned short bf16_1(float x) {
#if USE_CVT_PK_BF16
  return (unsigned short)(pack_bf16x2(x, 0.0f) & 0xFFFFu);
#else
  return (unsigned short)bf16_rne_bits(x);
#endif
}

__device__ __forceinline__ void ld2u4(ABFrag& f, const unsigned* p0, const unsigned* p1) {
  uint4 a = *(const uint4*)p0;
  uint4 b = *(const uint4*)p1;
  f.u[0] = a.x; f.u[1] = a.y; f.u[2] = a.z; f.u[3] = a.w;
  f.u[4] = b.x; f.u[5] = b.y; f.u[6] = b.z; f.u[7] = b.w;
}
__device__ __forceinline__ void wmma_bf16(CFrag& c, const ABFrag& a, const ABFrag& b) {
  c.v = __builtin_amdgcn_wmma_f32_16x16x32_bf16(false, a.v, false, b.v,
                                                (short)0, c.v, false, false);
}

// --- CDNA5 async global->LDS copy (ASYNCcnt path), with sync fallback -------
#if defined(__has_builtin)
#  if __has_builtin(__builtin_amdgcn_global_load_async_to_lds_b128)
#    define USE_ASYNC_LDS 1
#  endif
#endif
#ifndef USE_ASYNC_LDS
#  define USE_ASYNC_LDS 0
#endif

__device__ __forceinline__ void copy16_g2l(const unsigned* g, unsigned* l) {
#if USE_ASYNC_LDS
  __builtin_amdgcn_global_load_async_to_lds_b128(
      (__attribute__((address_space(1))) v4i*)g,
      (__attribute__((address_space(3))) v4i*)l, 0, 0);
#else
  *(uint4*)l = *(const uint4*)g;
#endif
}
__device__ __forceinline__ void wait_async_copies() {
#if USE_ASYNC_LDS
#  if __has_builtin(__builtin_amdgcn_s_wait_asynccnt)
  __builtin_amdgcn_s_wait_asynccnt(0);
#  else
  asm volatile("s_wait_asynccnt 0x0" ::: "memory");
#  endif
#endif
}

// ---------------------------------------------------------------------------
// Kernel 1: out_bf16[which][M][128] = x[M][E] @ W[E][128], bf16 WMMA, f32 acc.
// 256 threads (8 waves), tile 128(M) x 128(N), K-step 32, double-buffered LDS.
// Wave grid 4x2: each wave owns a 32x64 sub-tile (2x4 C fragments).
// ---------------------------------------------------------------------------
__global__ __launch_bounds__(256) void qkv_proj_kernel(
    const float* __restrict__ x, const float* __restrict__ Wq,
    const float* __restrict__ Wk, const float* __restrict__ Wv,
    unsigned short* __restrict__ qkv_out, int M, int E)
{
  constexpr int D = 128;
  // As[b][m][j] packs x[row][k0+2j], x[row][k0+2j+1]  (pairs along K)
  // Wt[b][n][j] packs W[k0+2j][n],  W[k0+2j+1][n]
  __shared__ __align__(16) unsigned As[2][128][20];
  __shared__ __align__(16) unsigned Wt[2][128][20];

  const int which = blockIdx.y;
  const float* W = (which == 0) ? Wq : ((which == 1) ? Wk : Wv);
  unsigned short* out = qkv_out + (size_t)which * (size_t)M * D;

  const int tid  = threadIdx.x;
  const int lane = tid & 31;
  const int wave = tid >> 5;
  const int half = lane >> 4;
  const int l16  = lane & 15;
  const int wm   = wave >> 1;
  const int wn   = wave & 1;
  const int m0   = blockIdx.x * 128;

  float2 xs[8];
  float  w0[8], w1[8];

  auto prefetch = [&](int k0) {
#pragma unroll
    for (int i = 0; i < 8; ++i) {
      int t = tid + i * 256;
      int m = t >> 4, j = t & 15;
      xs[i] = *(const float2*)(x + (size_t)(m0 + m) * E + k0 + 2 * j);
      int jj = t >> 7, n = t & 127;
      w0[i] = W[(size_t)(k0 + 2 * jj) * D + n];
      w1[i] = W[(size_t)(k0 + 2 * jj + 1) * D + n];
    }
  };
  auto commit = [&](int buf) {
#pragma unroll
    for (int i = 0; i < 8; ++i) {
      int t = tid + i * 256;
      int m = t >> 4, j = t & 15;
      As[buf][m][j] = pack_bf16x2(xs[i].x, xs[i].y);
      int jj = t >> 7, n = t & 127;
      Wt[buf][n][jj] = pack_bf16x2(w0[i], w1[i]);
    }
  };

  CFrag acc[2][4];
#pragma unroll
  for (int mt = 0; mt < 2; ++mt)
#pragma unroll
    for (int nt = 0; nt < 4; ++nt)
#pragma unroll
      for (int r = 0; r < 8; ++r) acc[mt][nt].f[r] = 0.0f;

  prefetch(0);
  commit(0);
  __syncthreads();

  const int nsteps = E / 32;
  for (int s = 0; s < nsteps; ++s) {
    const int buf = s & 1;
    if (s + 1 < nsteps) prefetch((s + 1) * 32);   // overlaps with WMMA below

    // A fragments (16x32 bf16): reg r<4 -> K=8h+2r ; r>=4 -> K=16+8h+2(r-4)
    ABFrag a[2];
#pragma unroll
    for (int mt = 0; mt < 2; ++mt) {
      const unsigned* base = &As[buf][wm * 32 + mt * 16 + l16][0];
      ld2u4(a[mt], base + 4 * half, base + 8 + 4 * half);
    }
    // B fragments (32x16 bf16): reg r<4 -> K=8h+2r ; r>=4 -> K=16+8h+2(r-4)
#pragma unroll
    for (int nt = 0; nt < 4; ++nt) {
      ABFrag b;
      const unsigned* base = &Wt[buf][wn * 64 + nt * 16 + l16][0];
      ld2u4(b, base + 4 * half, base + 8 + 4 * half);
#pragma unroll
      for (int mt = 0; mt < 2; ++mt) wmma_bf16(acc[mt][nt], a[mt], b);
    }

    if (s + 1 < nsteps) commit(buf ^ 1);
    __syncthreads();
  }

  // C layout: reg r, lanes 0-15: (M=r, N=lane); lanes 16-31: (M=r+8, N=lane-16)
#pragma unroll
  for (int mt = 0; mt < 2; ++mt)
#pragma unroll
    for (int nt = 0; nt < 4; ++nt)
#pragma unroll
      for (int r = 0; r < 8; ++r) {
        int row = m0 + wm * 32 + mt * 16 + r + 8 * half;
        int col = wn * 64 + nt * 16 + l16;
        out[(size_t)row * D + col] = bf16_1(acc[mt][nt].f[r]);
      }
}

// ---------------------------------------------------------------------------
// Kernel 2: causal flash attention.  Grid: (T/128, B).  256 threads = 8 waves.
// Each wave owns 16 query rows; 64 keys per iteration; K/V double-buffered.
// ---------------------------------------------------------------------------
__global__ __launch_bounds__(256) void attn_kernel(
    const unsigned short* __restrict__ Qb, const unsigned short* __restrict__ Kb,
    const unsigned short* __restrict__ Vb, float* __restrict__ out, int T)
{
  constexpr int D = 128;
  constexpr float SCALE = 0.088388347648318447f;   // 1/sqrt(128)

  // Ks[b][key][j] : raw bf16-pair copy of K tile (j = d-pair 0..63)
  // Vpk[b][d][j]  : packs V[j0+2j][d], V[j0+2j+1][d] (pairs along key)
  // Ps[row][col]  : bf16 probabilities, per-wave 16-row strips
  __shared__ __align__(16) unsigned Ks[2][64][68];
  __shared__ __align__(16) unsigned Vpk[2][128][36];
  __shared__ __align__(16) unsigned short Ps[128][72];

  const int tid  = threadIdx.x;
  const int lane = tid & 31;
  const int wave = tid >> 5;
  const int half = lane >> 4;
  const int l16  = lane & 15;

  const int it = blockIdx.x;
  const int b  = blockIdx.y;
  const int i0 = it * 128;
  const int bT = b * T;
  const int qrow = i0 + wave * 16;

  const unsigned* Q32 = (const unsigned*)Qb;
  const unsigned* K32 = (const unsigned*)Kb;
  const unsigned* V32 = (const unsigned*)Vb;

  unsigned sva[8], svb[8];                 // V staging registers

  auto k_issue = [&](int j0, int buf) {    // async raw copy into LDS
#pragma unroll
    for (int i = 0; i < 4; ++i) {
      int t = tid + i * 256;
      int m = t >> 4, qj = (t & 15) * 4;
      copy16_g2l(K32 + (size_t)(bT + j0 + m) * 64 + qj, &Ks[buf][m][qj]);
    }
  };
  auto v_fetch = [&](int j0) {
#pragma unroll
    for (int i = 0; i < 8; ++i) {
      int t = tid + i * 256;
      int j = t >> 6, dp = t & 63;
      sva[i] = V32[(size_t)(bT + j0 + 2 * j) * 64 + dp];
      svb[i] = V32[(size_t)(bT + j0 + 2 * j + 1) * 64 + dp];
    }
  };
  auto v_commit = [&](int buf) {
#pragma unroll
    for (int i = 0; i < 8; ++i) {
      int t = tid + i * 256;
      int j = t >> 6, dp = t & 63;
      Vpk[buf][2 * dp][j]     = (sva[i] & 0xFFFFu) | (svb[i] << 16);
      Vpk[buf][2 * dp + 1][j] = (sva[i] >> 16) | (svb[i] & 0xFFFF0000u);
    }
  };

  // --- Q fragments kept in registers for the whole kernel ---
  ABFrag qf[4];
  {
    const unsigned* qr = Q32 + (size_t)(bT + qrow + l16) * 64;
#pragma unroll
    for (int kk = 0; kk < 4; ++kk)
      ld2u4(qf[kk], qr + kk * 16 + 4 * half, qr + kk * 16 + 8 + 4 * half);
  }

  CFrag o[8];
#pragma unroll
  for (int dt = 0; dt < 8; ++dt)
#pragma unroll
    for (int r = 0; r < 8; ++r) o[dt].f[r] = 0.0f;
  float m_run[8], l_run[8];
#pragma unroll
  for (int r = 0; r < 8; ++r) { m_run[r] = -1e30f; l_run[r] = 0.0f; }

  // --- prologue: stage tile 0 ---
  k_issue(0, 0);
  v_fetch(0);
  v_commit(0);
  wait_async_copies();
  __syncthreads();

  const int jt_end = 2 * it + 1;          // causal: keys up to i0+127
  for (int jt = 0; jt <= jt_end; ++jt) {
    const int j0  = jt * 64;
    const int buf = jt & 1;
    const bool pf = (jt < jt_end);
    if (pf) {                              // prefetch next tile during compute
      k_issue(j0 + 64, buf ^ 1);
      v_fetch(j0 + 64);
    }

    // --- S = Q K^T ---
    CFrag sc[4];
#pragma unroll
    for (int nt = 0; nt < 4; ++nt)
#pragma unroll
      for (int r = 0; r < 8; ++r) sc[nt].f[r] = 0.0f;
#pragma unroll
    for (int kk = 0; kk < 4; ++kk) {
#pragma unroll
      for (int nt = 0; nt < 4; ++nt) {
        ABFrag bk;
        const unsigned* base = &Ks[buf][nt * 16 + l16][0];
        ld2u4(bk, base + kk * 16 + 4 * half, base + kk * 16 + 8 + 4 * half);
        wmma_bf16(sc[nt], qf[kk], bk);
      }
    }

    // --- scale + causal mask (wave-uniform branch, per-element select) ---
    const bool need_mask = (j0 + 63) > qrow;
#pragma unroll
    for (int nt = 0; nt < 4; ++nt)
#pragma unroll
      for (int r = 0; r < 8; ++r) {
        float s = sc[nt].f[r] * SCALE;
        if (need_mask) {
          int key = j0 + nt * 16 + l16;
          int q   = qrow + r + 8 * half;
          s = (key > q) ? -1e30f : s;
        }
        sc[nt].f[r] = s;
      }

    // --- online softmax (row reductions within each 16-lane half) ---
    float alpha[8];
#pragma unroll
    for (int r = 0; r < 8; ++r) {
      float mx = fmaxf(fmaxf(sc[0].f[r], sc[1].f[r]),
                       fmaxf(sc[2].f[r], sc[3].f[r]));
#pragma unroll
      for (int s = 1; s < 16; s <<= 1) mx = fmaxf(mx, __shfl_xor(mx, s, 32));
      float mn = fmaxf(m_run[r], mx);
      alpha[r] = __expf(m_run[r] - mn);
      m_run[r] = mn;
      float rs = 0.0f;
#pragma unroll
      for (int nt = 0; nt < 4; ++nt) {
        float p = __expf(sc[nt].f[r] - mn);
        sc[nt].f[r] = p;
        rs += p;
      }
#pragma unroll
      for (int s = 1; s < 16; s <<= 1) rs += __shfl_xor(rs, s, 32);
      l_run[r] = l_run[r] * alpha[r] + rs;
    }

    // --- rescale O, stage P (C layout -> LDS -> A layout) ---
#pragma unroll
    for (int dt = 0; dt < 8; ++dt)
#pragma unroll
      for (int r = 0; r < 8; ++r) o[dt].f[r] *= alpha[r];
#pragma unroll
    for (int nt = 0; nt < 4; ++nt)
#pragma unroll
      for (int r = 0; r < 8; ++r) {
        int row = wave * 16 + r + 8 * half;
        int col = nt * 16 + l16;
        Ps[row][col] = bf16_1(sc[nt].f[r]);
      }
    __syncthreads();                       // Ps visible

    // --- O += P @ V ---
#pragma unroll
    for (int kk = 0; kk < 2; ++kk) {
      ABFrag pa;
      const unsigned* pr = (const unsigned*)&Ps[wave * 16 + l16][0];
      ld2u4(pa, pr + kk * 16 + 4 * half, pr + kk * 16 + 8 + 4 * half);
#pragma unroll
      for (int dt = 0; dt < 8; ++dt) {
        ABFrag bv;
        const unsigned* vb = &Vpk[buf][dt * 16 + l16][0];
        ld2u4(bv, vb + kk * 16 + 4 * half, vb + kk * 16 + 8 + 4 * half);
        wmma_bf16(o[dt], pa, bv);
      }
    }

    if (pf) v_commit(buf ^ 1);             // into back buffer, no readers yet
    wait_async_copies();                   // K tile for next iter landed
    __syncthreads();
  }

  // --- normalize and write f32 output ---
  float inv[8];
#pragma unroll
  for (int r = 0; r < 8; ++r) inv[r] = 1.0f / l_run[r];
  const size_t obase = (size_t)(bT + qrow) * D;
#pragma unroll
  for (int dt = 0; dt < 8; ++dt)
#pragma unroll
    for (int r = 0; r < 8; ++r) {
      int qoff = r + 8 * half;
      out[obase + (size_t)qoff * D + dt * 16 + l16] = o[dt].f[r] * inv[r];
    }
}

// ---------------------------------------------------------------------------
extern "C" void kernel_launch(void* const* d_in, const int* in_sizes, int n_in,
                              void* d_out, int out_size, void* d_ws, size_t ws_size,
                              hipStream_t stream) {
  (void)n_in; (void)out_size; (void)ws_size; (void)in_sizes;
  const float* x  = (const float*)d_in[0];
  const float* Wq = (const float*)d_in[1];
  const float* Wk = (const float*)d_in[2];
  const float* Wv = (const float*)d_in[3];
  float* out = (float*)d_out;

  const int B = 4, T = 4096, E = 2048, D = 128;
  const int M = B * T;

  // Workspace: Q, K, V as bf16 [M][D] each (12 MB total).
  unsigned short* qkv = (unsigned short*)d_ws;
  const unsigned short* Qb = qkv;
  const unsigned short* Kb = qkv + (size_t)M * D;
  const unsigned short* Vb = qkv + 2 * (size_t)M * D;

  dim3 g1(M / 128, 3), blk(256);
  qkv_proj_kernel<<<g1, blk, 0, stream>>>(x, Wq, Wk, Wv, qkv, M, E);

  dim3 g2(T / 128, B);
  attn_kernel<<<g2, blk, 0, stream>>>(Qb, Kb, Vb, out, T);
}
